// HORSTCell_62173946577706
// MI455X (gfx1250) — matrix operand order repaired and checked by reading, not compiled, not run on hardware
//
#include <hip/hip_runtime.h>
#include <hip/hip_bf16.h>
#include <stdint.h>
#include <math.h>

typedef __attribute__((ext_vector_type(16))) __bf16 v16bf;
typedef __attribute__((ext_vector_type(8)))  float  v8f;

#define THREADS 256
#define CHW 131072   // 128 * 32 * 32
#define HW  1024

// ---------------------------------------------------------------------------
// helpers
// ---------------------------------------------------------------------------
__device__ inline uint32_t pack_bf16(float a, float b) {
  __bf16 ha = (__bf16)a, hb = (__bf16)b;
  unsigned short ua = __builtin_bit_cast(unsigned short, ha);
  unsigned short ub = __builtin_bit_cast(unsigned short, hb);
  return (uint32_t)ua | ((uint32_t)ub << 16);
}

// generic pointer -> 32-bit LDS byte offset (for async-to-LDS VDST operand)
__device__ inline uint32_t lds_addr32(const void* p) {
  return (uint32_t)(uintptr_t)(__attribute__((address_space(3))) const void*)p;
}

__device__ inline float block_reduce_sum(float v) {
  __shared__ float sb[256];
  int t = threadIdx.x;
  sb[t] = v;
  __syncthreads();
  for (int st = 128; st > 0; st >>= 1) {
    if (t < st) sb[t] += sb[t + st];
    __syncthreads();
  }
  float r = sb[0];
  __syncthreads();
  return r;
}

// ---------------------------------------------------------------------------
// Repack OIHW fp32 weights -> [kk][ci/2][co] packed bf16-pair u32
// (co contiguous -> coalesced global loads & clean async-to-LDS copies)
// ---------------------------------------------------------------------------
__global__ void wprep_kernel(const float* __restrict__ w, uint32_t* __restrict__ wtp,
                             int Cin, int KK) {
  const int cp_n = Cin >> 1;
  int total = 128 * cp_n * KK;
  for (int i = blockIdx.x * blockDim.x + threadIdx.x; i < total;
       i += gridDim.x * blockDim.x) {
    int co   = i & 127;
    int rest = i >> 7;
    int cp   = rest % cp_n;
    int kk   = rest / cp_n;
    float a = w[(size_t)co * Cin * KK + (2 * cp) * KK + kk];
    float b = w[(size_t)co * Cin * KK + (2 * cp + 1) * KK + kk];
    wtp[((size_t)kk * cp_n + cp) * 128 + co] = pack_bf16(a, b);
  }
}

// ---------------------------------------------------------------------------
// pos_emb norm clip: pe_n = pe * min(1, 1/max(||pe_s||, 1e-7))
// ---------------------------------------------------------------------------
__global__ __launch_bounds__(256)
void pe_norm_kernel(const float* __restrict__ pe, float* __restrict__ pe_n) {
  __shared__ float red[256];
  int t = threadIdx.x;
  for (int s = 0; s < 8; ++s) {
    float v = pe[s * 256 + t];
    red[t] = v * v;
    __syncthreads();
    for (int st = 128; st > 0; st >>= 1) {
      if (t < st) red[t] += red[t + st];
      __syncthreads();
    }
    float nrm = sqrtf(red[0]);
    float sc  = fminf(1.f, 1.f / fmaxf(nrm, 1e-7f));
    pe_n[s * 256 + t] = v * sc;
    __syncthreads();
  }
}

// ---------------------------------------------------------------------------
// Implicit-GEMM conv (Cout = 128, H = W = 32, NCHW fp32 in/out, bf16 WMMA).
// Virtual channel-concat of in0 (C0) and in1 (C1); optional per-(img/div,ch)
// bias (pos_emb fusion). Double-buffered LDS pipeline:
//   - B tile (pre-packed bf16 weights): async global->LDS scatter
//     (GLOBAL_LOAD_ASYNC_TO_LDS_B32, ASYNCcnt) issued for step s+1
//   - A tile: register-staged (f32 load + pad + bias + cvt) for step s+1
//   - 8x v_wmma_f32_16x16x32_bf16 on step s
// Block = 256 thr (8 wave32); grid.x = n_images * 8 (128-pixel M tiles).
// ---------------------------------------------------------------------------
__global__ __launch_bounds__(256)
void conv_wmma_kernel(const float* __restrict__ in0, int C0,
                      const float* __restrict__ in1, int C1,
                      const uint32_t* __restrict__ wtp, // [kk][Cin/2][128]
                      float* __restrict__ out,
                      const float* __restrict__ bias, // [img/bias_div][Cin] | null
                      int bias_div, int ks, int pad) {
  const int Cin    = C0 + C1;
  const int cbn    = Cin >> 5;          // 32-channel K-steps per tap
  const int KK     = ks * ks;
  const int nsteps = KK * cbn;
  const int img    = blockIdx.x >> 3;
  const int pbase  = (blockIdx.x & 7) * 128;
  const int tid    = threadIdx.x;
  const int lane   = tid & 31;
  const int wv     = tid >> 5;
  const int mbase  = wv * 16;

  __shared__ __bf16 As[2][128][40];   // 128 pixels x 32 K (+pad), x2 buffers
  __shared__ __bf16 Bs[2][128][40];   // 128 cout  x 32 K (+pad), x2 buffers

  v8f acc[8];
#pragma unroll
  for (int t = 0; t < 8; ++t)
#pragma unroll
    for (int j = 0; j < 8; ++j) acc[t][j] = 0.f;

  const float* bias_row =
      bias ? (bias + (size_t)(img / bias_div) * Cin) : (const float*)nullptr;

  // ---- async copy of B tile for step s into LDS buffer `buf` ----
  auto issueB = [&](int s, int buf) {
    int kk  = s / cbn;
    int cp0 = (s - kk * cbn) * 16;  // 16 channel-pairs per step
    const uint32_t* wrow = wtp + (size_t)kk * (Cin >> 1) * 128;
#pragma unroll
    for (int i = 0; i < 8; ++i) {
      int idx = i * 256 + tid;      // 2048 u32 per tile
      int co  = idx & 127;
      int cp  = idx >> 7;           // 0..15
      const uint32_t* g = wrow + (size_t)(cp0 + cp) * 128 + co;
      uint32_t ldsa = lds_addr32(&Bs[buf][co][2 * cp]);
      asm volatile("global_load_async_to_lds_b32 %0, %1, off"
                   :: "v"(ldsa), "v"((unsigned long long)(uintptr_t)g)
                   : "memory");
    }
  };

  // ---- stage A tile for step s into registers (pad + bias + gather) ----
  auto loadA = [&](int s, float* ar) {
    int kk = s / cbn;
    int cb = (s - kk * cbn) * 32;
    int ky = kk / ks - pad;
    int kx = kk % ks - pad;
#pragma unroll
    for (int i = 0; i < 8; ++i) {
      int idx = i * 256 + tid;      // 2048 channel-pairs
      int m   = idx & 127;
      int cp  = idx >> 7;
      int c   = cb + 2 * cp;        // even; pair never straddles C0 (C0 even)
      int p   = pbase + m;
      int y   = (p >> 5) + ky;
      int x   = (p & 31) + kx;
      float v0 = 0.f, v1 = 0.f;
      if ((unsigned)y < 32u && (unsigned)x < 32u) {
        const float* src = (c < C0)
            ? in0 + ((size_t)img * C0 + c) * HW
            : in1 + ((size_t)img * C1 + (c - C0)) * HW;
        int o = y * 32 + x;
        v0 = src[o];
        v1 = src[o + HW];
        if (bias_row) { v0 += bias_row[c]; v1 += bias_row[c + 1]; }
      }
      ar[2 * i]     = v0;
      ar[2 * i + 1] = v1;
    }
  };

  auto storeA = [&](const float* ar, int buf) {
#pragma unroll
    for (int i = 0; i < 8; ++i) {
      int idx = i * 256 + tid;
      int m   = idx & 127;
      int cp  = idx >> 7;
      *(uint32_t*)&As[buf][m][2 * cp] = pack_bf16(ar[2 * i], ar[2 * i + 1]);
    }
  };

  // ---- prologue: fill buffer 0 ----
  float ar[16];
  loadA(0, ar);
  issueB(0, 0);
  storeA(ar, 0);
  asm volatile("s_wait_asynccnt 0x0" ::: "memory");
  __syncthreads();

  int cur = 0;
  for (int s = 0; s < nsteps; ++s) {
    const bool more = (s + 1 < nsteps);
    if (more) {
      issueB(s + 1, cur ^ 1);   // async weights -> LDS, overlaps compute
      loadA(s + 1, ar);         // activations -> regs, overlaps compute
    }

    // ---- A fragment (ISA 16-bit A 16x32 layout) ----
    v16bf a;
    {
      int r   = lane & 15;
      int kof = (lane < 16) ? 0 : 8;
      const __bf16* ap = &As[cur][mbase + r][0];
#pragma unroll
      for (int j = 0; j < 8; ++j) {
        a[j]     = ap[kof + j];
        a[8 + j] = ap[16 + kof + j];
      }
    }
    // ---- 8 N-tiles: B fragment (ISA B 32x16 layout) + WMMA ----
#pragma unroll
    for (int t = 0; t < 8; ++t) {
      v16bf bfrag;
      int col = t * 16 + (lane & 15);
      int kof = (lane < 16) ? 0 : 8;
      const __bf16* bp = &Bs[cur][col][kof];
#pragma unroll
      for (int j = 0; j < 8; ++j) { bfrag[j] = bp[j]; bfrag[8 + j] = bp[16 + j]; }
      acc[t] = __builtin_amdgcn_wmma_f32_16x16x32_bf16(
          false, a, false, bfrag, (short)0, acc[t], false, false);
    }

    if (more) storeA(ar, cur ^ 1);
    asm volatile("s_wait_asynccnt 0x0" ::: "memory");
    __syncthreads();
    cur ^= 1;
  }

  // ---- writeback (ISA 32-bit C/D 16x16 layout) ----
  const int nloc = lane & 15;
  const int mof  = (lane < 16) ? 0 : 8;
#pragma unroll
  for (int t = 0; t < 8; ++t) {
#pragma unroll
    for (int r = 0; r < 8; ++r) {
      int co = t * 16 + nloc;
      int p  = pbase + mbase + mof + r;
      out[((size_t)img * 128 + co) * HW + p] = acc[t][r];
    }
  }
}

// ---------------------------------------------------------------------------
// GroupNorm(1 group): per-image mean / rsqrt(var+eps)
// ---------------------------------------------------------------------------
__global__ __launch_bounds__(256)
void gn_stats_kernel(const float* __restrict__ x, float* __restrict__ stats,
                     int nelem) {
  const size_t base = (size_t)blockIdx.x * nelem;
  float s = 0.f, s2 = 0.f;
  for (int i = threadIdx.x; i < nelem; i += 256) {
    float v = x[base + i];
    s += v; s2 += v * v;
  }
  s  = block_reduce_sum(s);
  s2 = block_reduce_sum(s2);
  if (threadIdx.x == 0) {
    float mu  = s / nelem;
    float var = s2 / nelem - mu * mu;
    stats[blockIdx.x * 2]     = mu;
    stats[blockIdx.x * 2 + 1] = rsqrtf(var + 1e-5f);
  }
}

__global__ void gn_apply_kernel(const float* __restrict__ x, float* __restrict__ y,
                                const float* __restrict__ stats,
                                const float* __restrict__ gw,
                                const float* __restrict__ gb, int total, int silu) {
  int i = blockIdx.x * blockDim.x + threadIdx.x;
  if (i >= total) return;
  int img = i >> 17;         // / CHW
  int c   = (i >> 10) & 127;
  float v = (x[i] - stats[img * 2]) * stats[img * 2 + 1] * gw[c] + gb[c];
  if (silu) v = v / (1.f + __expf(-v));
  y[i] = v;
}

// ---------------------------------------------------------------------------
// spatial_filter: sigmoid(conv3x3(concat(mean_c, max_c))) fused per image
// ---------------------------------------------------------------------------
__global__ __launch_bounds__(256)
void spatial_filter_kernel(const float* __restrict__ x, const float* __restrict__ w,
                           const float* __restrict__ bconv, float* __restrict__ sa) {
  __shared__ float pm[HW];
  __shared__ float px[HW];
  const int img = blockIdx.x;
  const float* xi = x + (size_t)img * CHW;
  for (int p = threadIdx.x; p < HW; p += 256) {
    float s = 0.f, mx = -3.4e38f;
    for (int c = 0; c < 128; ++c) {
      float v = xi[c * HW + p];
      s += v; mx = fmaxf(mx, v);
    }
    pm[p] = s * (1.f / 128.f);
    px[p] = mx;
  }
  __syncthreads();
  const float bb = bconv[0];
  for (int p = threadIdx.x; p < HW; p += 256) {
    int y = p >> 5, xx = p & 31;
    float acc = bb;
    for (int ky = 0; ky < 3; ++ky)
      for (int kx = 0; kx < 3; ++kx) {
        int iy = y + ky - 1, ix = xx + kx - 1;
        if ((unsigned)iy < 32u && (unsigned)ix < 32u) {
          int q = iy * 32 + ix;
          acc += w[ky * 3 + kx] * pm[q] + w[9 + ky * 3 + kx] * px[q];
        }
      }
    sa[(size_t)img * HW + p] = 1.f / (1.f + __expf(-acc));
  }
}

// q_fg[sb][c] = mean_p( k_sa[sb][p] * q[b][c][p] )
__global__ __launch_bounds__(256)
void qfg_kernel(const float* __restrict__ q, const float* __restrict__ k_sa,
                float* __restrict__ q_fg, int B) {
  int c  = blockIdx.x & 127;
  int sb = blockIdx.x >> 7;
  int b  = sb % B;
  const float* qp = q + ((size_t)b * 128 + c) * HW;
  const float* kp = k_sa + (size_t)sb * HW;
  float s = 0.f;
  for (int p = threadIdx.x; p < HW; p += 256) s += qp[p] * kp[p];
  s = block_reduce_sum(s);
  if (threadIdx.x == 0) q_fg[sb * 128 + c] = s * (1.f / 1024.f);
}

// s_prob[sb][p] = sigmoid( sum_c q_fg[sb][c] * k[sb][c][p] )
__global__ __launch_bounds__(256)
void sprob_kernel(const float* __restrict__ k, const float* __restrict__ q_fg,
                  float* __restrict__ s_prob) {
  __shared__ float qf[128];
  const int sb = blockIdx.x;
  if (threadIdx.x < 128) qf[threadIdx.x] = q_fg[sb * 128 + threadIdx.x];
  __syncthreads();
  const float* kp = k + (size_t)sb * CHW;
  for (int p = threadIdx.x; p < HW; p += 256) {
    float a = 0.f;
    for (int c = 0; c < 128; ++c) a += qf[c] * kp[c * HW + p];
    s_prob[(size_t)sb * HW + p] = 1.f / (1.f + __expf(-a));
  }
}

// t_log[sb] = scale * dot( q*q_sa (b), k*k_sa (sb) )
__global__ __launch_bounds__(256)
void tlogit_kernel(const float* __restrict__ q, const float* __restrict__ q_sa,
                   const float* __restrict__ k, const float* __restrict__ k_sa,
                   float* __restrict__ t_log, int B) {
  int sb = blockIdx.x, b = sb % B;
  const float* qp = q + (size_t)b * CHW;
  const float* qs = q_sa + (size_t)b * HW;
  const float* kp = k + (size_t)sb * CHW;
  const float* ks = k_sa + (size_t)sb * HW;
  float s = 0.f;
  for (int i = threadIdx.x; i < CHW; i += 256) {
    int p = i & 1023;
    s += qp[i] * qs[p] * kp[i] * ks[p];
  }
  s = block_reduce_sum(s);
  if (threadIdx.x == 0) t_log[sb] = s * 2.7621358640e-3f;  // 131072^-0.5
}

__global__ void softmax_s_kernel(const float* __restrict__ lg,
                                 float* __restrict__ pr, int B) {
  int b = threadIdx.x;
  if (b >= B) return;
  float mx = -3.4e38f;
  for (int s = 0; s < 8; ++s) mx = fmaxf(mx, lg[s * B + b]);
  float e[8], sum = 0.f;
  for (int s = 0; s < 8; ++s) { e[s] = __expf(lg[s * B + b] - mx); sum += e[s]; }
  float inv = 1.f / sum;
  for (int s = 0; s < 8; ++s) pr[s * B + b] = e[s] * inv;
}

// av[b][c][p] = sum_s t_prob[s][b] * s_prob[sb][p] * v[sb][c][p]
__global__ __launch_bounds__(256)
void av_kernel(const float* __restrict__ v, const float* __restrict__ s_prob,
               const float* __restrict__ t_prob, float* __restrict__ av, int B) {
  __shared__ float tp[8];
  const int bc = blockIdx.x, b = bc >> 7, c = bc & 127;
  if (threadIdx.x < 8) tp[threadIdx.x] = t_prob[threadIdx.x * B + b];
  __syncthreads();
  for (int p = threadIdx.x; p < HW; p += 256) {
    float a = 0.f;
    for (int s = 0; s < 8; ++s) {
      int sb = s * B + b;
      a += tp[s] * s_prob[(size_t)sb * HW + p] * v[((size_t)sb * 128 + c) * HW + p];
    }
    av[((size_t)b * 128 + c) * HW + p] = a;
  }
}

__global__ void newout_kernel(const float* __restrict__ att,
                              const unsigned char* __restrict__ mask,
                              float* __restrict__ y, int total) {
  int i = blockIdx.x * blockDim.x + threadIdx.x;
  if (i >= total) return;
  y[i] = mask[i >> 17] ? att[i] : 0.f;
}

__global__ void add_kernel(const float* __restrict__ a, const float* __restrict__ b,
                           float* __restrict__ y, int total) {
  int i = blockIdx.x * blockDim.x + threadIdx.x;
  if (i >= total) return;
  y[i] = a[i] + b[i];
}

__global__ __launch_bounds__(256)
void chmean_kernel(const float* __restrict__ x, float* __restrict__ g0) {
  const size_t base = (size_t)blockIdx.x * HW;
  float s = 0.f;
  for (int p = threadIdx.x; p < HW; p += 256) s += x[base + p];
  s = block_reduce_sum(s);
  if (threadIdx.x == 0) g0[blockIdx.x] = s * (1.f / 1024.f);
}

__global__ __launch_bounds__(256)
void se_kernel(const float* __restrict__ g0, const float* __restrict__ w1,
               const float* __restrict__ b1, const float* __restrict__ w2,
               const float* __restrict__ b2, float* __restrict__ g2) {
  __shared__ float g0s[128];
  __shared__ float g1s[32];
  const int b = blockIdx.x;
  if (threadIdx.x < 128) g0s[threadIdx.x] = g0[b * 128 + threadIdx.x];
  __syncthreads();
  if (threadIdx.x < 32) {
    float a = b1[threadIdx.x];
    for (int j = 0; j < 128; ++j) a += w1[threadIdx.x * 128 + j] * g0s[j];
    g1s[threadIdx.x] = fmaxf(a, 0.f);
  }
  __syncthreads();
  if (threadIdx.x < 128) {
    float a = b2[threadIdx.x];
    for (int j = 0; j < 32; ++j) a += w2[threadIdx.x * 32 + j] * g1s[j];
    g2[b * 128 + threadIdx.x] = 1.f / (1.f + __expf(-a));
  }
}

__global__ void outputs_kernel(const float* __restrict__ o, const float* __restrict__ g2,
                               float* __restrict__ y, int total) {
  int i = blockIdx.x * blockDim.x + threadIdx.x;
  if (i >= total) return;
  int b = i >> 17, c = (i >> 10) & 127;
  y[i] = o[i] * (1.f + g2[b * 128 + c]);
}

// ---------------------------------------------------------------------------
extern "C" void kernel_launch(void* const* d_in, const int* in_sizes, int n_in,
                              void* d_out, int out_size, void* d_ws, size_t ws_size,
                              hipStream_t stream) {
  (void)in_sizes; (void)n_in; (void)out_size; (void)ws_size;
  const int B = 16, SB = 128;

  const float* inputs = (const float*)d_in[0];
  const float* hq     = (const float*)d_in[1];
  const float* oq     = (const float*)d_in[2];
  const unsigned char* mask = (const unsigned char*)d_in[3];
  const float* enc_w  = (const float*)d_in[4];
  const float* enc_gw = (const float*)d_in[5];
  const float* enc_gb = (const float*)d_in[6];
  const float* qpre_w = (const float*)d_in[7];
  const float* qpre_gw= (const float*)d_in[8];
  const float* qpre_gb= (const float*)d_in[9];
  const float* kpre_w = (const float*)d_in[10];
  const float* kpre_gw= (const float*)d_in[11];
  const float* kpre_gb= (const float*)d_in[12];
  const float* vpre_w = (const float*)d_in[13];
  const float* vpre_gw= (const float*)d_in[14];
  const float* vpre_gb= (const float*)d_in[15];
  const float* qsa_w  = (const float*)d_in[16];
  const float* qsa_b  = (const float*)d_in[17];
  const float* ksa_w  = (const float*)d_in[18];
  const float* ksa_b  = (const float*)d_in[19];
  const float* vpost_w = (const float*)d_in[20];
  const float* vpost_gw= (const float*)d_in[21];
  const float* vpost_gb= (const float*)d_in[22];
  const float* pos_emb = (const float*)d_in[23];
  const float* outt_w  = (const float*)d_in[24];
  const float* outt_gw = (const float*)d_in[25];
  const float* outt_gb = (const float*)d_in[26];
  const float* se_w1 = (const float*)d_in[27];
  const float* se_b1 = (const float*)d_in[28];
  const float* se_w2 = (const float*)d_in[29];
  const float* se_b2 = (const float*)d_in[30];

  float* out_outputs = (float*)d_out;
  float* out_ht      = out_outputs + (size_t)B * CHW;
  float* out_new     = out_ht + (size_t)B * CHW;

  // --- workspace carve-out ---
  char* wsb = (char*)d_ws;
  size_t off = 0;
  auto alloc = [&](size_t bytes) -> void* {
    void* p = wsb + off;
    off += (bytes + 255) & ~(size_t)255;
    return p;
  };
  uint32_t* wt_enc   = (uint32_t*)alloc((size_t)128 * 64  * 9 * 2);
  uint32_t* wt_qpre  = (uint32_t*)alloc((size_t)128 * 128 * 9 * 2);
  uint32_t* wt_kpre  = (uint32_t*)alloc((size_t)128 * 256 * 9 * 2);
  uint32_t* wt_vpre  = (uint32_t*)alloc((size_t)128 * 256 * 9 * 2);
  uint32_t* wt_vpost = (uint32_t*)alloc((size_t)128 * 128 * 1 * 2);
  uint32_t* wt_outt  = (uint32_t*)alloc((size_t)128 * 192 * 9 * 2);
  float* pe_n   = (float*)alloc(8 * 256 * 4);
  float* stats  = (float*)alloc(2 * SB * 4);
  float* bigA   = (float*)alloc((size_t)SB * CHW * 4);  // enc tmp, then v
  float* bigB   = (float*)alloc((size_t)SB * CHW * 4);  // k
  float* qbuf   = (float*)alloc((size_t)B * CHW * 4);
  float* avbuf  = (float*)alloc((size_t)B * CHW * 4);   // av, then o
  float* attbuf = (float*)alloc((size_t)B * CHW * 4);
  float* ahbuf  = (float*)alloc((size_t)B * CHW * 4);
  float* q_sa   = (float*)alloc((size_t)B * HW * 4);
  float* k_sa   = (float*)alloc((size_t)SB * HW * 4);
  float* q_fg   = (float*)alloc((size_t)SB * 128 * 4);
  float* s_prob = (float*)alloc((size_t)SB * HW * 4);
  float* t_log  = (float*)alloc(SB * 4);
  float* t_prob = (float*)alloc(SB * 4);
  float* g0     = (float*)alloc((size_t)B * 128 * 4);
  float* g2     = (float*)alloc((size_t)B * 128 * 4);

  const int totB  = B * CHW;
  const int ewB   = (totB + 255) / 256;
  const int totSB = SB * CHW;
  const int ewSB  = (totSB + 255) / 256;

  // --- weight repack + pos_emb normalize ---
  wprep_kernel<<<144, 256, 0, stream>>>(enc_w,  wt_enc,  64, 9);
  wprep_kernel<<<288, 256, 0, stream>>>(qpre_w, wt_qpre, 128, 9);
  wprep_kernel<<<576, 256, 0, stream>>>(kpre_w, wt_kpre, 256, 9);
  wprep_kernel<<<576, 256, 0, stream>>>(vpre_w, wt_vpre, 256, 9);
  wprep_kernel<<<32, 256, 0, stream>>>(vpost_w, wt_vpost, 128, 1);
  wprep_kernel<<<432, 256, 0, stream>>>(outt_w, wt_outt, 192, 9);
  pe_norm_kernel<<<1, 256, 0, stream>>>(pos_emb, pe_n);

  // --- ht = silu(gn(conv_enc(inputs))) -> directly into d_out slot ---
  conv_wmma_kernel<<<B * 8, 256, 0, stream>>>(inputs, 64, nullptr, 0, wt_enc,
                                              bigA, nullptr, 1, 3, 1);
  gn_stats_kernel<<<B, 256, 0, stream>>>(bigA, stats, CHW);
  gn_apply_kernel<<<ewB, 256, 0, stream>>>(bigA, out_ht, stats, enc_gw, enc_gb, totB, 1);

  // --- q = gn(conv_qpre(ht)) ---
  conv_wmma_kernel<<<B * 8, 256, 0, stream>>>(out_ht, 128, nullptr, 0, wt_qpre,
                                              qbuf, nullptr, 1, 3, 1);
  gn_stats_kernel<<<B, 256, 0, stream>>>(qbuf, stats, CHW);
  gn_apply_kernel<<<ewB, 256, 0, stream>>>(qbuf, qbuf, stats, qpre_gw, qpre_gb, totB, 0);

  // --- k = gn(conv_kpre(concat(hq, oq)))  (virtual concat) ---
  conv_wmma_kernel<<<SB * 8, 256, 0, stream>>>(hq, 128, oq, 128, wt_kpre,
                                               bigB, nullptr, 1, 3, 1);
  gn_stats_kernel<<<SB, 256, 0, stream>>>(bigB, stats, CHW);
  gn_apply_kernel<<<ewSB, 256, 0, stream>>>(bigB, bigB, stats, kpre_gw, kpre_gb, totSB, 0);

  // --- v = gn(conv_vpre(concat(hq, oq) + pe))  (pe fused as bias) ---
  conv_wmma_kernel<<<SB * 8, 256, 0, stream>>>(hq, 128, oq, 128, wt_vpre,
                                               bigA, pe_n, B, 3, 1);
  gn_stats_kernel<<<SB, 256, 0, stream>>>(bigA, stats, CHW);
  gn_apply_kernel<<<ewSB, 256, 0, stream>>>(bigA, bigA, stats, vpre_gw, vpre_gb, totSB, 0);

  // --- spatial filters ---
  spatial_filter_kernel<<<B, 256, 0, stream>>>(qbuf, qsa_w, qsa_b, q_sa);
  spatial_filter_kernel<<<SB, 256, 0, stream>>>(bigB, ksa_w, ksa_b, k_sa);

  // --- attention scalars ---
  qfg_kernel<<<SB * 128, 256, 0, stream>>>(qbuf, k_sa, q_fg, B);
  sprob_kernel<<<SB, 256, 0, stream>>>(bigB, q_fg, s_prob);
  tlogit_kernel<<<SB, 256, 0, stream>>>(qbuf, q_sa, bigB, k_sa, t_log, B);
  softmax_s_kernel<<<1, 32, 0, stream>>>(t_log, t_prob, B);

  // --- av = sum_s t_prob * (s_prob * v)   (s_prob gating fused) ---
  av_kernel<<<B * 128, 256, 0, stream>>>(bigA, s_prob, t_prob, avbuf, B);

  // --- att = silu(gn(conv1x1_vpost(av))) ---
  conv_wmma_kernel<<<B * 8, 256, 0, stream>>>(avbuf, 128, nullptr, 0, wt_vpost,
                                              attbuf, nullptr, 1, 1, 0);
  gn_stats_kernel<<<B, 256, 0, stream>>>(attbuf, stats, CHW);
  gn_apply_kernel<<<ewB, 256, 0, stream>>>(attbuf, attbuf, stats, vpost_gw, vpost_gb, totB, 1);

  // --- new_out, ah = att + ht ---
  newout_kernel<<<ewB, 256, 0, stream>>>(attbuf, mask, out_new, totB);
  add_kernel<<<ewB, 256, 0, stream>>>(attbuf, out_ht, ahbuf, totB);

  // --- o = silu(gn(conv_outt(concat(inputs, ah)))) (virtual concat) ---
  conv_wmma_kernel<<<B * 8, 256, 0, stream>>>(inputs, 64, ahbuf, 128, wt_outt,
                                              avbuf, nullptr, 1, 3, 1);
  gn_stats_kernel<<<B, 256, 0, stream>>>(avbuf, stats, CHW);
  gn_apply_kernel<<<ewB, 256, 0, stream>>>(avbuf, avbuf, stats, outt_gw, outt_gb, totB, 1);

  // --- SE block + outputs = o * (1 + g) ---
  chmean_kernel<<<B * 128, 256, 0, stream>>>(avbuf, g0);
  se_kernel<<<B, 256, 0, stream>>>(g0, se_w1, se_b1, se_w2, se_b2, g2);
  outputs_kernel<<<ewB, 256, 0, stream>>>(avbuf, g2, out_outputs, totB);
}
